// FirstOrderODENet_87574383165784
// MI455X (gfx1250) — compile-verified, matrix operand order checked by reading, not run on hardware
//
#include <hip/hip_runtime.h>
#include <hip/hip_bf16.h>
#include <stdint.h>

typedef __bf16 bf16_t;
typedef __attribute__((ext_vector_type(16))) __bf16 v16bf;
typedef __attribute__((ext_vector_type(8)))  float  v8f;

#define DIMD   1024          // input dim  (x, k columns)
#define DIMH   4096          // hidden dim
#define BROWS  32            // batch rows per workgroup
#define THREADS 512          // 16 waves (wave32)
#define HCHUNK 512           // hidden-dim slab per fused pass
#define NCHUNK (DIMH / HCHUNK)
#define NSTEPS 64
#define WSZ    4194304       // elements per weight matrix (4096*1024)
#define HSLAB  (BROWS * HCHUNK)   // elements per h slab

union FragU { v16bf v; uint4 q[2]; };

__device__ __forceinline__ bf16_t to_bf16(float f) {
  unsigned u = __float_as_uint(f);
  u += 0x7FFFu + ((u >> 16) & 1u);           // round-to-nearest-even
  unsigned short s = (unsigned short)(u >> 16);
  return __builtin_bit_cast(bf16_t, s);
}

__device__ __forceinline__ float fast_tanh(float x) {
#if __has_builtin(__builtin_amdgcn_tanhf)
  return __builtin_amdgcn_tanhf(x);          // V_TANH_F32 (CDNA5 trans unit)
#else
  return tanhf(x);
#endif
}

// Build a 16x16x32 bf16 WMMA A/B fragment from a row-major bf16 row.
// Lane-half h holds K = {k0+8h .. k0+8h+7} and {k0+16+8h .. +7} (ISA 7.12.2).
__device__ __forceinline__ v16bf load_frag(const bf16_t* rowptr, int k0, int half) {
  const uint4* p = (const uint4*)(rowptr + k0 + 8 * half);
  FragU f;
  f.q[0] = p[0];     // 8 bf16: K = k0+8h .. +7
  f.q[1] = p[2];     // 8 bf16: K = k0+16+8h .. +7
  return f.v;
}

// ---- fused RHS: yacc = decoder(tanh(encoder(x_eval))) for this WG's 32 rows.
// wave -> (rt = wave>>3 row-tile, wg8 = wave&7 column group)
__device__ void rhs_eval(const bf16_t* __restrict__ x_eval,   // [32][1024] bf16 LDS
                         bf16_t* __restrict__ h_base,         // 2x [32][512] bf16 LDS
                         const bf16_t* __restrict__ W1b,      // [4096][1024]
                         const bf16_t* __restrict__ W2b,      // [1024][4096]
                         const float* __restrict__ b1,
                         const float* __restrict__ b2,
                         int rt, int wg8, int ln, int lh, v8f (&yacc)[8]) {
  #pragma unroll
  for (int t = 0; t < 8; ++t) {
    const float bv = b2[wg8 * 128 + t * 16 + ln];
    #pragma unroll
    for (int v = 0; v < 8; ++v) yacc[t][v] = bv;
  }

  for (int c = 0; c < NCHUNK; ++c) {
    const int hbase = c * HCHUNK;
    bf16_t* h_chk = h_base + (c & 1) * HSLAB;   // double buffer: 1 barrier/chunk
    v8f hacc[4];
    #pragma unroll
    for (int t = 0; t < 4; ++t) {
      #pragma unroll
      for (int v = 0; v < 8; ++v) hacc[t][v] = 0.0f;
    }

    // GEMM1 slab: h_chunk = x @ W1^T  over k = 0..1023
    for (int k0 = 0; k0 < DIMD; k0 += 32) {
      const v16bf a = load_frag(x_eval + (rt * 16 + ln) * DIMD, k0, lh);
      #pragma unroll
      for (int t = 0; t < 4; ++t) {
        const int hcol = hbase + wg8 * 64 + t * 16 + ln;       // W1 row = output col
        const v16bf b = load_frag(W1b + (size_t)hcol * DIMD, k0, lh);
        hacc[t] = __builtin_amdgcn_wmma_f32_16x16x32_bf16(
            false, a, false, b, (short)0, hacc[t], false, false);
      }
    }

    // epilogue: bias + tanh -> bf16 slab in LDS (buffer c&1; readers of this
    // buffer finished chunk c-2's GEMM2 before the chunk c-1 barrier)
    #pragma unroll
    for (int t = 0; t < 4; ++t) {
      const int ccol = wg8 * 64 + t * 16 + ln;
      const float bv = b1[hbase + ccol];
      #pragma unroll
      for (int v = 0; v < 8; ++v) {
        const int m = rt * 16 + v + 8 * lh;                    // C/D layout: M = v + 8*lanehalf
        h_chk[m * HCHUNK + ccol] = to_bf16(fast_tanh(hacc[t][v] + bv));
      }
    }
    __syncthreads();   // slab visible to all waves

    // GEMM2 partial: yacc += h_slab @ W2^T  over k = hbase..hbase+511
    for (int k0 = 0; k0 < HCHUNK; k0 += 32) {
      const v16bf a = load_frag(h_chk + (rt * 16 + ln) * HCHUNK, k0, lh);
      #pragma unroll
      for (int t = 0; t < 8; ++t) {
        const int dcol = wg8 * 128 + t * 16 + ln;              // W2 row = output col
        const v16bf b = load_frag(W2b + (size_t)dcol * DIMH + hbase, k0, lh);
        yacc[t] = __builtin_amdgcn_wmma_f32_16x16x32_bf16(
            false, a, false, b, (short)0, yacc[t], false, false);
      }
    }
  }
}

// ---- RK4 stage update; all elements are lane-exclusive (WMMA C/D ownership).
template <int STAGE>
__device__ __forceinline__ void stage_update(float* __restrict__ x_cur,
                                             bf16_t* __restrict__ x_eval,
                                             v8f (&yacc)[8], v8f (&accum)[8],
                                             int rt, int wg8, int ln, int lh) {
  const float dt = 1.0f / (float)NSTEPS;
  #pragma unroll
  for (int t = 0; t < 8; ++t) {
    const int col = wg8 * 128 + t * 16 + ln;
    #pragma unroll
    for (int v = 0; v < 8; ++v) {
      const int idx = (rt * 16 + v + 8 * lh) * DIMD + col;
      const float k  = yacc[t][v];
      const float xc = x_cur[idx];
      float xe;
      if (STAGE == 0)      { accum[t][v] = k;          xe = xc + 0.5f * dt * k; }
      else if (STAGE == 1) { accum[t][v] += 2.0f * k;  xe = xc + 0.5f * dt * k; }
      else if (STAGE == 2) { accum[t][v] += 2.0f * k;  xe = xc + dt * k; }
      else { const float xn = xc + (dt / 6.0f) * (accum[t][v] + k);
             x_cur[idx] = xn; xe = xn; }
      x_eval[idx] = to_bf16(xe);
    }
  }
  __syncthreads();   // x_eval visible before next RHS
}

__global__ __launch_bounds__(THREADS, 1) void odenet_persistent(
    const float* __restrict__ x_in, const float* __restrict__ b1,
    const float* __restrict__ b2, const bf16_t* __restrict__ W1b,
    const bf16_t* __restrict__ W2b, float* __restrict__ out) {
  extern __shared__ __attribute__((aligned(16))) char smem[];
  float*  x_cur  = (float*)smem;                           // 32*1024*4 = 128 KB
  bf16_t* x_eval = (bf16_t*)(smem + 131072);               // 32*1024*2 =  64 KB
  bf16_t* h_base = (bf16_t*)(smem + 131072 + 65536);       // 2x 32*512*2 = 64 KB

  const int tid  = threadIdx.x;
  const int wave = tid >> 5;
  const int lane = tid & 31;
  const int ln   = lane & 15;
  const int lh   = lane >> 4;
  const int rt   = wave >> 3;    // row-tile 0/1
  const int wg8  = wave & 7;     // column group
  const size_t row0 = (size_t)blockIdx.x * BROWS;

  // load this WG's 32 rows of x into LDS (f32 state + bf16 eval copy)
  const float4* xin4 = (const float4*)(x_in + row0 * DIMD);
  float4* xc4 = (float4*)x_cur;
  for (int i = tid; i < BROWS * DIMD / 4; i += THREADS) {
    const float4 v = xin4[i];
    xc4[i] = v;
    const int e = i * 4;
    x_eval[e + 0] = to_bf16(v.x);
    x_eval[e + 1] = to_bf16(v.y);
    x_eval[e + 2] = to_bf16(v.z);
    x_eval[e + 3] = to_bf16(v.w);
  }
  __syncthreads();

  v8f accum[8];
  v8f yacc[8];
  for (int step = 0; step < NSTEPS; ++step) {
    rhs_eval(x_eval, h_base, W1b, W2b, b1, b2, rt, wg8, ln, lh, yacc);
    stage_update<0>(x_cur, x_eval, yacc, accum, rt, wg8, ln, lh);
    rhs_eval(x_eval, h_base, W1b, W2b, b1, b2, rt, wg8, ln, lh, yacc);
    stage_update<1>(x_cur, x_eval, yacc, accum, rt, wg8, ln, lh);
    rhs_eval(x_eval, h_base, W1b, W2b, b1, b2, rt, wg8, ln, lh, yacc);
    stage_update<2>(x_cur, x_eval, yacc, accum, rt, wg8, ln, lh);
    rhs_eval(x_eval, h_base, W1b, W2b, b1, b2, rt, wg8, ln, lh, yacc);
    stage_update<3>(x_cur, x_eval, yacc, accum, rt, wg8, ln, lh);
  }

  float4* out4 = (float4*)(out + row0 * DIMD);
  for (int i = tid; i < BROWS * DIMD / 4; i += THREADS)
    out4[i] = xc4[i];
}

// f32 -> bf16 (RNE) weight conversion; W1 and W2 have identical element counts.
__global__ __launch_bounds__(256) void convert_weights(
    const float* __restrict__ W1, const float* __restrict__ W2,
    bf16_t* __restrict__ W1b, bf16_t* __restrict__ W2b) {
  const size_t i = (size_t)blockIdx.x * 256 + threadIdx.x;
  if (i < (size_t)WSZ) {
    W1b[i] = to_bf16(W1[i]);
    W2b[i] = to_bf16(W2[i]);
  }
}

extern "C" void kernel_launch(void* const* d_in, const int* in_sizes, int n_in,
                              void* d_out, int out_size, void* d_ws, size_t ws_size,
                              hipStream_t stream) {
  (void)in_sizes; (void)n_in; (void)out_size; (void)ws_size;
  const float* x  = (const float*)d_in[0];
  const float* W1 = (const float*)d_in[1];   // [4096][1024]
  const float* b1 = (const float*)d_in[2];   // [4096]
  const float* W2 = (const float*)d_in[3];   // [1024][4096]
  const float* b2 = (const float*)d_in[4];   // [1024]
  float* out = (float*)d_out;

  bf16_t* W1b = (bf16_t*)d_ws;
  bf16_t* W2b = (bf16_t*)((char*)d_ws + (size_t)WSZ * sizeof(bf16_t));

  convert_weights<<<WSZ / 256, 256, 0, stream>>>(W1, W2, W1b, W2b);

  const int lds = 131072 + 65536 + 65536;    // 256 KB of the WGP's 320 KB
  hipFuncSetAttribute(reinterpret_cast<const void*>(odenet_persistent),
                      hipFuncAttributeMaxDynamicSharedMemorySize, lds);
  odenet_persistent<<<4096 / BROWS, THREADS, lds, stream>>>(x, b1, b2, W1b, W2b, out);
}
